// MistralAttention_KIVI_56564719289141
// MI455X (gfx1250) — compile-verified, matrix operand order checked by reading, not run on hardware
//
#include <hip/hip_runtime.h>

#define DEV __device__ __forceinline__

typedef __attribute__((ext_vector_type(16))) _Float16 v16h;
typedef __attribute__((ext_vector_type(8)))  _Float16 v8h;
typedef __attribute__((ext_vector_type(8)))  float    v8f;

constexpr int B = 4, QL = 8, HID = 4096;
constexpr int NH = 32, KVH = 8, HD = 128, G = NH / KVH;
constexpr int GS = 64, SQ = 4096, SF = 64;
constexpr int KV_LEN = SQ + SF + QL;   // 4168
constexpr int SFQ = SF + QL;           // 72
constexpr int CHUNK = 32;
constexpr int NCHUNK = (KV_LEN + CHUNK - 1) / CHUNK; // 131
constexpr int SPLIT = 4;               // KV-range splits per attention tile
constexpr int NTILE = B * KVH * 2;     // 64 (b, kvh, mtile) attention tiles
constexpr float RSCALE = 0.08838834764831845f;        // 1/sqrt(HD)

// K-offset of fragment element pair j for this lane-half (16-bit A/B layout:
// lanes 0-15 hold K {0..7,16..23}, lanes 16-31 hold K {8..15,24..31}).
DEV int frag_koff(int j, int hi) {
  return ((j & 4) ? 16 : 0) + hi * 8 + (j & 3) * 2;
}

DEV v8f wmma_f16(v16h a, v16h b, v8f c) {
  return __builtin_amdgcn_wmma_f32_16x16x32_f16(false, a, false, b, (short)0, c,
                                                false, false);
}

// Load one 32-wide K-slice fragment from contiguous f32 memory with four
// global_load_b128: frag elements 0..7 = p[hi*8 .. +8), 8..15 = p[16+hi*8..).
DEV v16h load_frag(const float* p, int hi) {
  const float4 f0 = *(const float4*)(p + hi * 8);
  const float4 f1 = *(const float4*)(p + hi * 8 + 4);
  const float4 f2 = *(const float4*)(p + 16 + hi * 8);
  const float4 f3 = *(const float4*)(p + 16 + hi * 8 + 4);
  v16h r;
  r[0]  = (_Float16)f0.x; r[1]  = (_Float16)f0.y;
  r[2]  = (_Float16)f0.z; r[3]  = (_Float16)f0.w;
  r[4]  = (_Float16)f1.x; r[5]  = (_Float16)f1.y;
  r[6]  = (_Float16)f1.z; r[7]  = (_Float16)f1.w;
  r[8]  = (_Float16)f2.x; r[9]  = (_Float16)f2.y;
  r[10] = (_Float16)f2.z; r[11] = (_Float16)f2.w;
  r[12] = (_Float16)f3.x; r[13] = (_Float16)f3.y;
  r[14] = (_Float16)f3.z; r[15] = (_Float16)f3.w;
  return r;
}

DEV v16h load_frag_scaled(const float* p, int hi, float sc) {
  const float4 f0 = *(const float4*)(p + hi * 8);
  const float4 f1 = *(const float4*)(p + hi * 8 + 4);
  const float4 f2 = *(const float4*)(p + 16 + hi * 8);
  const float4 f3 = *(const float4*)(p + 16 + hi * 8 + 4);
  v16h r;
  r[0]  = (_Float16)(f0.x * sc); r[1]  = (_Float16)(f0.y * sc);
  r[2]  = (_Float16)(f0.z * sc); r[3]  = (_Float16)(f0.w * sc);
  r[4]  = (_Float16)(f1.x * sc); r[5]  = (_Float16)(f1.y * sc);
  r[6]  = (_Float16)(f1.z * sc); r[7]  = (_Float16)(f1.w * sc);
  r[8]  = (_Float16)(f2.x * sc); r[9]  = (_Float16)(f2.y * sc);
  r[10] = (_Float16)(f2.z * sc); r[11] = (_Float16)(f2.w * sc);
  r[12] = (_Float16)(f3.x * sc); r[13] = (_Float16)(f3.y * sc);
  r[14] = (_Float16)(f3.z * sc); r[15] = (_Float16)(f3.w * sc);
  return r;
}

// ---------------------------------------------------------------------------
// Stage 1: fused QKV projection. A = hidden [32,4096], B = [wq|wk|wv]^T
// column-blocks. One 16x16 output tile per wave, f16 WMMA over K, all
// fragment traffic as b128 loads.
// ---------------------------------------------------------------------------
__global__ __launch_bounds__(256) void qkv_gemm_kernel(
    const float* __restrict__ hs, const float* __restrict__ wq,
    const float* __restrict__ wk, const float* __restrict__ wv,
    float* __restrict__ qbuf, float* __restrict__ kext,
    float* __restrict__ vext) {
  const int lane = threadIdx.x & 31;
  const int wave = threadIdx.x >> 5;
  const int NTN = (HID + 2 * KVH * HD) / 16; // 384 n-tiles
  const int tile = blockIdx.x * 8 + wave;
  if (tile >= 2 * NTN) return;
  const int mtile = tile / NTN;
  const int ntile = tile % NTN;
  const int ln = lane & 15, hi = lane >> 4;

  const float* arow = hs + (size_t)(mtile * 16 + ln) * HID;
  const int cl = ntile * 16 + ln; // this lane's output feature
  const float* brow;
  if (cl < HID)                 brow = wq + (size_t)cl * HID;
  else if (cl < HID + KVH * HD) brow = wk + (size_t)(cl - HID) * HID;
  else                          brow = wv + (size_t)(cl - HID - KVH * HD) * HID;

  v8f acc = {};
  for (int k0 = 0; k0 < HID; k0 += 32) {
    __builtin_prefetch(arow + k0 + 256, 0, 0);
    __builtin_prefetch(brow + k0 + 256, 0, 0);
    const v16h a   = load_frag(arow + k0, hi);
    const v16h bfr = load_frag(brow + k0, hi);
    acc = wmma_f16(a, bfr, acc);
  }

#pragma unroll
  for (int i = 0; i < 8; ++i) {
    const int r = mtile * 16 + i + hi * 8; // global row = b*QL + q
    const int bb = r >> 3, q = r & 7;
    const int c = ntile * 16 + ln;
    const float v = acc[i];
    if (c < HID) {
      const int hh = c >> 7, d = c & 127;
      qbuf[(((size_t)bb * NH + hh) * QL + q) * HD + d] = v;
    } else if (c < HID + KVH * HD) {
      const int cc = c - HID;
      const int kh = cc >> 7, d = cc & 127;
      kext[(((size_t)bb * KVH + kh) * SFQ + SF + q) * HD + d] = v;
    } else {
      const int cc = c - HID - KVH * HD;
      const int kh = cc >> 7, d = cc & 127;
      vext[(((size_t)bb * KVH + kh) * SFQ + SF + q) * HD + d] = v;
    }
  }
}

// ---------------------------------------------------------------------------
// Stage 2a: copy full-precision residual caches (float4 granularity).
// ---------------------------------------------------------------------------
__global__ __launch_bounds__(256) void copy_full_kernel(
    const float4* __restrict__ kf, const float4* __restrict__ vf,
    float4* __restrict__ kext, float4* __restrict__ vext) {
  const int HD4 = HD / 4;                 // 32
  const int n = B * KVH * SF * HD4;       // 65536 float4s
  for (int i = blockIdx.x * blockDim.x + threadIdx.x; i < n;
       i += gridDim.x * blockDim.x) {
    const int d4 = i % HD4;
    const int s = (i / HD4) % SF;
    const int bk = i / (HD4 * SF);
    const size_t dst = ((size_t)bk * SFQ + s) * HD4 + d4;
    kext[dst] = kf[i];
    vext[dst] = vf[i];
  }
}

// ---------------------------------------------------------------------------
// Stage 2b: RoPE on q (all heads) and new k rows (HF half-rotate convention).
// ---------------------------------------------------------------------------
__global__ __launch_bounds__(256) void rope_kernel(float* __restrict__ qbuf,
                                                   float* __restrict__ kext,
                                                   const int* __restrict__ pos) {
  const int nq = B * NH * QL * (HD / 2);  // 65536
  const int nk = B * KVH * QL * (HD / 2); // 16384
  const int i = blockIdx.x * blockDim.x + threadIdx.x;
  float* base;
  int d, q, bb;
  if (i < nq) {
    d = i % 64; q = (i / 64) % QL;
    const int bh = i / (64 * QL);
    bb = bh / NH;
    base = qbuf + ((size_t)bh * QL + q) * HD;
  } else if (i < nq + nk) {
    const int j = i - nq;
    d = j % 64; q = (j / 64) % QL;
    const int bk = j / (64 * QL);
    bb = bk / KVH;
    base = kext + ((size_t)bk * SFQ + SF + q) * HD;
  } else {
    return;
  }
  const float p = (float)pos[bb * QL + q];
  const float invf = __powf(10000.0f, -(float)(2 * d) / (float)HD);
  const float ang = p * invf;
  const float c = __cosf(ang), s = __sinf(ang);
  const float x1 = base[d], x2 = base[d + 64];
  base[d]      = x1 * c - x2 * s;
  base[d + 64] = x2 * c + x1 * s;
}

// ---------------------------------------------------------------------------
// Stage 3: flash attention over 32-key chunks with fused 2-bit dequant.
// Grid: (b, kvh, mtile, split) -> 256 blocks of 8 waves. Waves stride the
// chunk range; per-block partial (M, l, O) states are merged in LDS
// (ds_add_f32) and written to workspace for the finalize pass.
// ---------------------------------------------------------------------------
__global__ __launch_bounds__(256) void flash_kernel(
    const float* __restrict__ qbuf, const int* __restrict__ kq,
    const float* __restrict__ ksc, const float* __restrict__ kmn,
    const float* __restrict__ kext, const int* __restrict__ vq,
    const float* __restrict__ vsc, const float* __restrict__ vmn,
    const float* __restrict__ vext, float* __restrict__ part_o,
    float* __restrict__ part_m, float* __restrict__ part_l) {
  const int split = blockIdx.x & (SPLIT - 1);
  const int tile_id = blockIdx.x >> 2;
  const int mtile = tile_id & 1;
  const int bk = tile_id >> 1;
  const int kvh = bk % KVH, bb = bk / KVH;
  const int lane = threadIdx.x & 31;
  const int wave = threadIdx.x >> 5;
  const int ln = lane & 15, hi = lane >> 4;

  __shared__ _Float16 p_lds[8][16][32]; // per-wave P tile (D-layout -> A-layout)
  __shared__ float m_sh[8][16], l_sh[8][16];
  __shared__ float o_sh[16][128];
  __shared__ float ls_sh[16];

  for (int i = threadIdx.x; i < 16 * 128; i += 256) (&o_sh[0][0])[i] = 0.f;
  if (threadIdx.x < 16) ls_sh[threadIdx.x] = 0.f;
  __syncthreads();

  // Q fragments for this 16-row tile, pre-scaled by 1/sqrt(HD). b128 loads.
  const int gr = mtile * 16 + ln;
  const int g = gr >> 3, q = gr & 7, h = kvh * G + g;
  const float* qrow = qbuf + (((size_t)bb * NH + h) * QL + q) * HD;
  v16h aq[4];
#pragma unroll
  for (int ks = 0; ks < 4; ++ks) aq[ks] = load_frag_scaled(qrow + ks * 32, hi, RSCALE);

  const size_t bkh = (size_t)bb * KVH + kvh;
  const int*   kqp = kq + bkh * HD * SQ;            // [HD][SQ] codes
  const float* ksp = ksc + bkh * HD * (SQ / GS);
  const float* kmp = kmn + bkh * HD * (SQ / GS);
  const float* kep = kext + bkh * SFQ * HD;
  const int*   vqp = vq + bkh * SQ * HD;            // [SQ][HD] codes
  const float* vsp = vsc + bkh * SQ * (HD / GS);
  const float* vmp = vmn + bkh * SQ * (HD / GS);
  const float* vep = vext + bkh * SFQ * HD;

  float mrun[8], lrun[8];
  v8f o[8];
  {
    v8f z = {};
#pragma unroll
    for (int i = 0; i < 8; ++i) { mrun[i] = -3.0e38f; lrun[i] = 0.f; o[i] = z; }
  }

  // 32 wave-streams (8 waves x 4 split blocks) stride the 131 chunks.
  for (int cc = split * 8 + wave; cc < NCHUNK; cc += SPLIT * 8) {
    const int s0 = cc * CHUNK;
    const bool quant = (s0 < SQ); // SQ is a multiple of CHUNK: no straddling
    // Hint the next chunk of this wave-stream into cache.
    if (s0 + SPLIT * 8 * CHUNK < SQ) {
      __builtin_prefetch(kqp + (size_t)(s0 + SPLIT * 8 * CHUNK) + ln, 0, 0);
      __builtin_prefetch(vqp + (size_t)(s0 + SPLIT * 8 * CHUNK) * HD + lane, 0, 0);
    }
    // Group index is wave-uniform: a 32-key chunk never straddles GS=64.
    const int grp = s0 / GS;
    v8f sc2[2];
#pragma unroll
    for (int t = 0; t < 2; ++t) {
      const int s = s0 + t * 16 + ln; // this lane's key (B-matrix column)
      v8f acc = {};
      if (quant) {
#pragma unroll
        for (int ks = 0; ks < 4; ++ks) {
          v16h bkf;
#pragma unroll
          for (int j = 0; j < 8; ++j) {
            const int d = ks * 32 + frag_koff(j, hi);
            const float v0 =
                (float)kqp[(size_t)d * SQ + s] * ksp[d * (SQ / GS) + grp] +
                kmp[d * (SQ / GS) + grp];
            const float v1 =
                (float)kqp[(size_t)(d + 1) * SQ + s] *
                    ksp[(d + 1) * (SQ / GS) + grp] +
                kmp[(d + 1) * (SQ / GS) + grp];
            bkf[2 * j] = (_Float16)v0;
            bkf[2 * j + 1] = (_Float16)v1;
          }
          acc = wmma_f16(aq[ks], bkf, acc);
        }
      } else {
        // Full-precision tail: clamp row (invalid keys are masked to -inf
        // below, so a clamped-garbage load is harmless) and use b128 loads.
        const int srow = (s < KV_LEN) ? (s - SQ) : (SFQ - 1);
        const float* kr = kep + (size_t)srow * HD;
#pragma unroll
        for (int ks = 0; ks < 4; ++ks)
          acc = wmma_f16(aq[ks], load_frag(kr + ks * 32, hi), acc);
      }
      // causal mask over the QL new tokens + out-of-range padding
#pragma unroll
      for (int i = 0; i < 8; ++i) {
        const int qi = (mtile * 16 + i + hi * 8) & 7;
        float x = acc[i];
        if (s >= KV_LEN || (s >= SQ + SF && (s - (SQ + SF)) > qi)) x = -1.0e30f;
        sc2[t][i] = x;
      }
    }

    // Online softmax update (rows live per-vgpr; 16-lane half reductions).
#pragma unroll
    for (int i = 0; i < 8; ++i) {
      float rm = fmaxf(sc2[0][i], sc2[1][i]);
      rm = fmaxf(rm, __shfl_xor(rm, 1, 32));
      rm = fmaxf(rm, __shfl_xor(rm, 2, 32));
      rm = fmaxf(rm, __shfl_xor(rm, 4, 32));
      rm = fmaxf(rm, __shfl_xor(rm, 8, 32));
      const float mnew = fmaxf(mrun[i], rm);
      const float alpha = __expf(mrun[i] - mnew);
      const float p0 = __expf(sc2[0][i] - mnew);
      const float p1 = __expf(sc2[1][i] - mnew);
      const int rt = i + hi * 8;
      p_lds[wave][rt][ln]      = (_Float16)p0;
      p_lds[wave][rt][16 + ln] = (_Float16)p1;
      float rs = p0 + p1;
      rs += __shfl_xor(rs, 1, 32);
      rs += __shfl_xor(rs, 2, 32);
      rs += __shfl_xor(rs, 4, 32);
      rs += __shfl_xor(rs, 8, 32);
      lrun[i] = lrun[i] * alpha + rs;
      mrun[i] = mnew;
#pragma unroll
      for (int t = 0; t < 8; ++t) o[t][i] *= alpha;
    }

    // Re-read P in A-layout: two ds_load_b128 per lane (DS ops are in-order).
    v16h ap;
    {
      const v8h pl0 = *(const v8h*)(&p_lds[wave][ln][hi * 8]);
      const v8h pl1 = *(const v8h*)(&p_lds[wave][ln][16 + hi * 8]);
#pragma unroll
      for (int e = 0; e < 8; ++e) { ap[e] = pl0[e]; ap[8 + e] = pl1[e]; }
    }

    // O += P @ V (8 n-tiles of HD), dequant fused into the B fragment.
#pragma unroll
    for (int t = 0; t < 8; ++t) {
      const int d = t * 16 + ln;
      v16h bvf;
#pragma unroll
      for (int j = 0; j < 8; ++j) {
        const int kk = frag_koff(j, hi);
        const int sA = s0 + kk;
        float v0, v1;
        if (quant) {
          const int gA = d >> 6;
          v0 = (float)vqp[(size_t)sA * HD + d] * vsp[sA * (HD / GS) + gA] +
               vmp[sA * (HD / GS) + gA];
          v1 = (float)vqp[(size_t)(sA + 1) * HD + d] *
                   vsp[(sA + 1) * (HD / GS) + gA] +
               vmp[(sA + 1) * (HD / GS) + gA];
        } else {
          // Invalid keys carry P == 0, so clamped loads are harmless.
          const int r0 = (sA < KV_LEN) ? (sA - SQ) : (SFQ - 1);
          const int r1 = (sA + 1 < KV_LEN) ? (sA + 1 - SQ) : (SFQ - 1);
          v0 = vep[(size_t)r0 * HD + d];
          v1 = vep[(size_t)r1 * HD + d];
        }
        bvf[2 * j] = (_Float16)v0;
        bvf[2 * j + 1] = (_Float16)v1;
      }
      o[t] = wmma_f16(ap, bvf, o[t]);
    }
  }

  // Merge the 8 per-wave partial (m, l, O) states within the block.
  if (ln == 0) {
#pragma unroll
    for (int i = 0; i < 8; ++i) {
      m_sh[wave][i + hi * 8] = mrun[i];
      l_sh[wave][i + hi * 8] = lrun[i];
    }
  }
  __syncthreads();
#pragma unroll
  for (int i = 0; i < 8; ++i) {
    const int rt = i + hi * 8;
    float M = -3.0e38f;
#pragma unroll
    for (int w = 0; w < 8; ++w) M = fmaxf(M, m_sh[w][rt]);
    const float f = __expf(mrun[i] - M);
#pragma unroll
    for (int t = 0; t < 8; ++t) atomicAdd(&o_sh[rt][t * 16 + ln], f * o[t][i]);
    if (ln == 0) atomicAdd(&ls_sh[rt], f * lrun[i]);
  }
  __syncthreads();

  // Publish this block's partial (unnormalized O, block max M, partial l).
  const size_t pidx = (size_t)tile_id * SPLIT + split;
  float4* po = (float4*)(part_o + pidx * (16 * 128));
  const float4* os4 = (const float4*)(&o_sh[0][0]);
  for (int idx = threadIdx.x; idx < 16 * 128 / 4; idx += 256) po[idx] = os4[idx];
  if (threadIdx.x < 16) {
    float M = -3.0e38f;
#pragma unroll
    for (int w = 0; w < 8; ++w) M = fmaxf(M, m_sh[w][threadIdx.x]);
    part_m[pidx * 16 + threadIdx.x] = M;
    part_l[pidx * 16 + threadIdx.x] = ls_sh[threadIdx.x];
  }
}

// ---------------------------------------------------------------------------
// Stage 3b: merge the SPLIT partial softmax states and normalize.
// ---------------------------------------------------------------------------
__global__ __launch_bounds__(256) void finalize_kernel(
    const float* __restrict__ part_o, const float* __restrict__ part_m,
    const float* __restrict__ part_l, float* __restrict__ attnbuf) {
  const int tile_id = blockIdx.x; // 0..63
  const int mtile = tile_id & 1;
  const int bk = tile_id >> 1;
  const int kvh = bk % KVH, bb = bk / KVH;
  for (int idx = threadIdx.x; idx < 16 * 128; idx += 256) {
    const int rt = idx >> 7, d = idx & 127;
    float M = -3.0e38f;
#pragma unroll
    for (int s = 0; s < SPLIT; ++s)
      M = fmaxf(M, part_m[((size_t)tile_id * SPLIT + s) * 16 + rt]);
    float num = 0.f, den = 0.f;
#pragma unroll
    for (int s = 0; s < SPLIT; ++s) {
      const size_t pidx = (size_t)tile_id * SPLIT + s;
      const float f = __expf(part_m[pidx * 16 + rt] - M);
      num += f * part_o[pidx * (16 * 128) + idx];
      den += f * part_l[pidx * 16 + rt];
    }
    const int gr2 = mtile * 16 + rt;
    const int g2 = gr2 >> 3, q2 = gr2 & 7, h2 = kvh * G + g2;
    attnbuf[(((size_t)bb * NH + h2) * QL + q2) * HD + d] = num / den;
  }
}

// ---------------------------------------------------------------------------
// Stage 4: output projection [32,4096] x wo^T -> d_out (b128 fragment loads).
// ---------------------------------------------------------------------------
__global__ __launch_bounds__(256) void out_proj_kernel(
    const float* __restrict__ attnbuf, const float* __restrict__ wo,
    float* __restrict__ out) {
  const int lane = threadIdx.x & 31;
  const int wave = threadIdx.x >> 5;
  const int NTN = HID / 16; // 256
  const int tile = blockIdx.x * 8 + wave;
  if (tile >= 2 * NTN) return;
  const int mtile = tile / NTN, ntile = tile % NTN;
  const int ln = lane & 15, hi = lane >> 4;
  const int r = mtile * 16 + ln; // row = b*QL + q
  const int bb = r >> 3, q = r & 7;
  const float* brow = wo + (size_t)(ntile * 16 + ln) * HID;

  v8f acc = {};
  for (int k0 = 0; k0 < HID; k0 += 32) {
    __builtin_prefetch(brow + k0 + 256, 0, 0);
    // The 32-wide K-slice [k0, k0+32) stays inside one head (128-aligned).
    const int hh = k0 >> 7, dd = k0 & 127;
    const float* ab = attnbuf + (((size_t)bb * NH + hh) * QL + q) * HD + dd;
    const v16h a   = load_frag(ab, hi);
    const v16h bfr = load_frag(brow + k0, hi);
    acc = wmma_f16(a, bfr, acc);
  }
#pragma unroll
  for (int i = 0; i < 8; ++i) {
    const int rr = mtile * 16 + i + hi * 8;
    out[(size_t)rr * HID + ntile * 16 + ln] = acc[i];
  }
}

// ---------------------------------------------------------------------------
extern "C" void kernel_launch(void* const* d_in, const int* in_sizes, int n_in,
                              void* d_out, int out_size, void* d_ws,
                              size_t ws_size, hipStream_t stream) {
  (void)in_sizes; (void)n_in; (void)out_size; (void)ws_size;
  const float* hs      = (const float*)d_in[0];
  const float* wq      = (const float*)d_in[1];
  const float* wk      = (const float*)d_in[2];
  const float* wv      = (const float*)d_in[3];
  const float* wo      = (const float*)d_in[4];
  const float* k_scale = (const float*)d_in[5];
  const float* k_mn    = (const float*)d_in[6];
  const float* k_full  = (const float*)d_in[7];
  const float* v_scale = (const float*)d_in[8];
  const float* v_mn    = (const float*)d_in[9];
  const float* v_full  = (const float*)d_in[10];
  const int*   k_quant = (const int*)d_in[11];
  const int*   v_quant = (const int*)d_in[12];
  const int*   pos_ids = (const int*)d_in[13];
  // d_in[14] (attention_mask) is reconstructed analytically (causal tail).
  float* out = (float*)d_out;

  float* qbuf = (float*)d_ws;                         // 131072 f
  float* kext = qbuf + (size_t)B * NH * QL * HD;      // 294912 f
  float* vext = kext + (size_t)B * KVH * SFQ * HD;    // 294912 f
  float* attnbuf = vext + (size_t)B * KVH * SFQ * HD; // 131072 f
  float* part_o = attnbuf + (size_t)B * NH * QL * HD; // 64*4*2048 = 524288 f
  float* part_m = part_o + (size_t)NTILE * SPLIT * 16 * 128; // 4096 f
  float* part_l = part_m + (size_t)NTILE * SPLIT * 16;       // 4096 f

  qkv_gemm_kernel<<<96, 256, 0, stream>>>(hs, wq, wk, wv, qbuf, kext, vext);
  copy_full_kernel<<<256, 256, 0, stream>>>(
      (const float4*)k_full, (const float4*)v_full, (float4*)kext,
      (float4*)vext);
  rope_kernel<<<(B * NH * QL * (HD / 2) + B * KVH * QL * (HD / 2) + 255) / 256,
                256, 0, stream>>>(qbuf, kext, pos_ids);
  flash_kernel<<<NTILE * SPLIT, 256, 0, stream>>>(
      qbuf, k_quant, k_scale, k_mn, kext, v_quant, v_scale, v_mn, vext,
      part_o, part_m, part_l);
  finalize_kernel<<<NTILE, 256, 0, stream>>>(part_o, part_m, part_l, attnbuf);
  out_proj_kernel<<<64, 256, 0, stream>>>(attnbuf, wo, out);
}
